// ColumnSelfAttention_42082089566795
// MI455X (gfx1250) — matrix operand order repaired and checked by reading, not compiled, not run on hardware
//
#include <hip/hip_runtime.h>

#define RR 128
#define CC 1024
#define EE 768
#define HH 12
#define DD 64
#define QSCALE 0.125f   // 64^-0.5

typedef __attribute__((ext_vector_type(16))) __bf16 v16bf;
typedef __attribute__((ext_vector_type(8)))  float  v8f;
typedef __attribute__((ext_vector_type(4)))  unsigned int u32x4;
typedef int v4i __attribute__((vector_size(4 * sizeof(int))));
typedef __attribute__((address_space(1))) v4i gv4i;   // global (printed __device__)
typedef __attribute__((address_space(3))) v4i lv4i;   // LDS (printed __shared__)

union FragU { v16bf v; u32x4 u[2]; };

// ---- CDNA5 async global->LDS copy (ASYNCcnt-tracked), with sync fallback ----
#if defined(__gfx1250__) && __has_builtin(__builtin_amdgcn_global_load_async_to_lds_b128)
#define ASYNC_LDS 1
#else
#define ASYNC_LDS 0
#endif

__device__ inline void cp128(__bf16* lds, const __bf16* g) {
#if ASYNC_LDS
  __builtin_amdgcn_global_load_async_to_lds_b128((gv4i*)g, (lv4i*)lds, 0, 0);
#else
  *(u32x4*)lds = *(const u32x4*)g;   // sync fallback: reg round-trip, DScnt-tracked
#endif
}
__device__ inline void wait_cp() {
#if ASYNC_LDS
#if __has_builtin(__builtin_amdgcn_s_wait_asynccnt)
  __builtin_amdgcn_s_wait_asynccnt(0);
#else
  asm volatile("s_wait_asynccnt 0x0" ::: "memory");
#endif
#endif
}

// 32-bit *byte* offset addressing: base + zext(i32 byteoff) folds into GVS
// addressing (SGPR64 base + VGPR32 offset + IOFFSET).
__device__ inline const u32x4* gp16(const __bf16* base, unsigned elemOff) {
  return (const u32x4*)((const char*)base + elemOff * 2u);
}
__device__ inline void st_bf16(__bf16* base, unsigned elemOff, float v) {
  *(__bf16*)((char*)base + elemOff * 2u) = (__bf16)v;
}
__device__ inline void st_f32(float* base, unsigned elemOff, float v) {
  *(float*)((char*)base + elemOff * 4u) = v;
}

// A-matrix 16x32 bf16 fragment (ISA 7.12.2): lane m = lane&15, half = lane>>4,
// K = {8*half+0..7, 16+8*half+0..7} -> two contiguous 16-byte runs per lane.
__device__ inline v16bf load_fragA(const __bf16* base, unsigned rowStride, unsigned m0, unsigned k0) {
  const unsigned lane = threadIdx.x & 31u;
  const unsigned off = (m0 + (lane & 15u)) * rowStride + k0 + 8u * (lane >> 4);
  FragU f;
  f.u[0] = *gp16(base, off);
  f.u[1] = *gp16(base, off + 16u);
  return f.v;
}

// B-matrix 32x16 bf16 fragment: lane n = lane&15, half = lane>>4,
// K = {8*half+0..7, 16+8*half+0..7} -> two contiguous 16-byte runs per lane.
// `base` stores B transposed: row n contiguous in K.
__device__ inline v16bf load_fragB(const __bf16* base, unsigned rowStride, unsigned n0, unsigned k0) {
  const unsigned lane = threadIdx.x & 31u;
  const unsigned off = (n0 + (lane & 15u)) * rowStride + k0 + 8u * (lane >> 4);
  FragU f;
  f.u[0] = *gp16(base, off);
  f.u[1] = *gp16(base, off + 16u);
  return f.v;
}

__device__ inline v8f wmma_bf16(v16bf a, v16bf b, v8f c) {
  return __builtin_amdgcn_wmma_f32_16x16x32_bf16(false, a, false, b, (short)0, c, false, false);
}

// ---------------- conversion kernels ----------------
__global__ void cvt_f32_bf16(const float* __restrict__ s, __bf16* __restrict__ d, long long n) {
  long long i = (long long)blockIdx.x * blockDim.x + threadIdx.x;
  const long long stride = (long long)gridDim.x * blockDim.x;
  for (; i < n; i += stride) d[i] = (__bf16)s[i];
}

__global__ void wt_transpose(const float* __restrict__ W, __bf16* __restrict__ Wt) {
  int i = blockIdx.x * blockDim.x + threadIdx.x;
  if (i >= EE * EE) return;
  int k = i / EE, n = i % EE;                          // coalesced read of W[k][n]
  st_bf16(Wt, (unsigned)n * EE + (unsigned)k, W[i]);   // Wt[n][k]
}

// ---------------- fused QKV projection ----------------
// grid (RC/128, EE/64), block 256 (8 waves). Weight tiles (shared by all waves)
// are double-buffered in LDS via async copies; WMMAs read B from LDS (short
// ds latency) while the async engine prefetches the next K-step.
__global__ void __launch_bounds__(256) qkv_gemm(
    const __bf16* __restrict__ xb,
    const __bf16* __restrict__ wqt, const __bf16* __restrict__ wkt, const __bf16* __restrict__ wvt,
    const float* __restrict__ bq, const float* __restrict__ bk, const float* __restrict__ bv,
    __bf16* __restrict__ qb, __bf16* __restrict__ kb, __bf16* __restrict__ vb) {
  __shared__ __align__(16) __bf16 Bl[2][3][64][40];   // [buf][q/k/v][n][k] (+pad)
  const unsigned tid = threadIdx.x;
  const unsigned wave = tid >> 5, lane = tid & 31u;
  const unsigned m0 = blockIdx.x * 128u + wave * 16u;
  const unsigned n0 = blockIdx.y * 64u;
  const unsigned nl = tid >> 2;             // staging row 0..63
  const unsigned kc = (tid & 3u) * 8u;      // staging chunk 0/8/16/24
  const __bf16* wsrc[3] = {wqt, wkt, wvt};

  v8f aq[4] = {}, ak[4] = {}, av[4] = {};
#pragma unroll
  for (unsigned p = 0; p < 3; ++p)
    cp128(&Bl[0][p][nl][kc], wsrc[p] + (n0 + nl) * EE + kc);
  wait_cp();
  __syncthreads();

  for (unsigned step = 0; step < EE / 32u; ++step) {
    const unsigned buf = step & 1u;
    if (step + 1 < EE / 32u) {
      const unsigned knext = (step + 1) * 32u;
#pragma unroll
      for (unsigned p = 0; p < 3; ++p)
        cp128(&Bl[buf ^ 1u][p][nl][kc], wsrc[p] + (n0 + nl) * EE + knext + kc);
    }
    v16bf a = load_fragA(xb, EE, m0, step * 32u);
#pragma unroll
    for (unsigned s = 0; s < 4; ++s) {
      v16bf fq = load_fragB(&Bl[buf][0][0][0], 40, s * 16u, 0);
      v16bf fk = load_fragB(&Bl[buf][1][0][0], 40, s * 16u, 0);
      v16bf fv = load_fragB(&Bl[buf][2][0][0], 40, s * 16u, 0);
      aq[s] = wmma_bf16(a, fq, aq[s]);
      ak[s] = wmma_bf16(a, fk, ak[s]);
      av[s] = wmma_bf16(a, fv, av[s]);
    }
    wait_cp();        // prefetch landed (this wave)
    __syncthreads();  // all waves done reading buf; next buffer visible
  }

  const unsigned hsel = lane >> 4, nlane = lane & 15u;
#pragma unroll
  for (unsigned s = 0; s < 4; ++s) {
    const unsigned col = n0 + s * 16u + nlane;
    const float bqv = bq[col], bkv = bk[col], bvv = bv[col];
#pragma unroll
    for (unsigned r = 0; r < 8; ++r) {
      const unsigned o = (m0 + r + 8u * hsel) * EE + col;
      st_bf16(qb, o, (aq[s][r] + bqv) * QSCALE);   // pre-scale q
      st_bf16(kb, o, ak[s][r] + bkv);
      st_bf16(vb, o, av[s][r] + bvv);
    }
  }
}

// ---------------- attention per (h, c) ----------------
// grid (CC, HH), block 256. K tile async-staged to LDS (shared by all waves),
// then the K region is reused for the P tile after S is consumed.
__global__ void __launch_bounds__(256) attn_kernel(
    const __bf16* __restrict__ qb, const __bf16* __restrict__ kb, const __bf16* __restrict__ vb,
    const float* __restrict__ mask, float* __restrict__ probs, __bf16* __restrict__ ctx) {
  __shared__ __align__(16) char smem[34816 + 17408];
  __bf16 (*Kl)[72]   = (__bf16(*)[72])smem;            // K tile [j][d], 18.4KB
  __bf16 (*Pld)[136] = (__bf16(*)[136])smem;           // P tile reuses K region
  __bf16 (*Vt)[136]  = (__bf16(*)[136])(smem + 34816); // V transposed [d][j]
  const unsigned c = blockIdx.x, h = blockIdx.y;
  const unsigned tid = threadIdx.x;
  const unsigned cbase = c * EE + h * DD;
  const unsigned rstride = CC * EE;

  // Stage K[j][d] -> LDS (async, 4 x b128 per thread)
#pragma unroll
  for (unsigned p = 0; p < 4; ++p) {
    const unsigned chunk = tid + 256u * p;
    const unsigned j = chunk >> 3, d8 = (chunk & 7u) * 8u;
    cp128(&Kl[j][d8], kb + cbase + j * rstride + d8);
  }
  // Stage V[j][d] -> Vt[d][j] (needs transpose: VGPR round-trip)
#pragma unroll
  for (unsigned pass = 0; pass < 4; ++pass) {
    const unsigned j = pass * 32u + (tid >> 3);
    const unsigned d0 = (tid & 7u) * 8u;
    u32x4 u = *gp16(vb, cbase + j * rstride + d0);
    const __bf16* ph = (const __bf16*)&u;
#pragma unroll
    for (unsigned e = 0; e < 8; ++e) Vt[d0 + e][j] = ph[e];
  }
  wait_cp();
  __syncthreads();

  const unsigned wave = tid >> 5, lane = tid & 31u;
  const unsigned hsel = lane >> 4, nlane = lane & 15u;
  const unsigned i0 = wave * 16u;

  // S = Q K^T (16x128 slab per wave); Q frags global, K frags from LDS
  v8f acc[8] = {};
#pragma unroll
  for (unsigned k0 = 0; k0 < DD; k0 += 32u) {
    v16bf a = load_fragA(qb + cbase, rstride, i0, k0);
    v16bf bk8[8];
#pragma unroll
    for (unsigned t = 0; t < 8; ++t) bk8[t] = load_fragB(&Kl[0][0], 72, t * 16u, k0);
#pragma unroll
    for (unsigned t = 0; t < 8; ++t) acc[t] = wmma_bf16(a, bk8[t], acc[t]);
  }

  // mask (query rows) + softmax over j via 16-lane shfl_xor reductions (wave32)
#pragma unroll
  for (unsigned r = 0; r < 8; ++r) {
    const unsigned i = i0 + r + 8u * hsel;
    const float m = mask[i * CC + c];
    const float keep = 1.0f - m;
    float mx = -3.4e38f;
#pragma unroll
    for (unsigned t = 0; t < 8; ++t) {
      const float s = acc[t][r] * keep - 10000.0f * m;
      acc[t][r] = s;
      mx = fmaxf(mx, s);
    }
    mx = fmaxf(mx, __shfl_xor(mx, 1, 32));
    mx = fmaxf(mx, __shfl_xor(mx, 2, 32));
    mx = fmaxf(mx, __shfl_xor(mx, 4, 32));
    mx = fmaxf(mx, __shfl_xor(mx, 8, 32));
    float sum = 0.0f;
#pragma unroll
    for (unsigned t = 0; t < 8; ++t) {
      const float e = __expf(acc[t][r] - mx);
      acc[t][r] = e;
      sum += e;
    }
    sum += __shfl_xor(sum, 1, 32);
    sum += __shfl_xor(sum, 2, 32);
    sum += __shfl_xor(sum, 4, 32);
    sum += __shfl_xor(sum, 8, 32);
    const float inv = 1.0f / sum;
#pragma unroll
    for (unsigned t = 0; t < 8; ++t) acc[t][r] *= inv;
  }

  __syncthreads();   // all waves done reading Kl before P overwrites it

  // write probs (fp32 output) + P tile (bf16) into reused LDS region
  const unsigned pbase = (h * CC + c) * (RR * RR);
#pragma unroll
  for (unsigned r = 0; r < 8; ++r) {
    const unsigned i = i0 + r + 8u * hsel;
#pragma unroll
    for (unsigned t = 0; t < 8; ++t) {
      const unsigned j = t * 16u + nlane;
      const float p = acc[t][r];
      st_f32(probs, pbase + i * RR + j, p);
      Pld[i][j] = (__bf16)p;
    }
  }
  __syncthreads();

  // ctx = P V (16x64 slab per wave, K-dim = 128), all operands in LDS
  v8f cacc[4] = {};
#pragma unroll
  for (unsigned j0 = 0; j0 < RR; j0 += 32u) {
    v16bf a = load_fragA(&Pld[0][0], 136, i0, j0);
    v16bf bv4[4];
#pragma unroll
    for (unsigned s = 0; s < 4; ++s) bv4[s] = load_fragB(&Vt[0][0], 136, s * 16u, j0);
#pragma unroll
    for (unsigned s = 0; s < 4; ++s) cacc[s] = wmma_bf16(a, bv4[s], cacc[s]);
  }
#pragma unroll
  for (unsigned s = 0; s < 4; ++s) {
#pragma unroll
    for (unsigned r = 0; r < 8; ++r) {
      const unsigned i = i0 + r + 8u * hsel;
      st_bf16(ctx, i * rstride + cbase + s * 16u + nlane, cacc[s][r]);
    }
  }
}

// ---------------- output projection ----------------
__global__ void __launch_bounds__(256) out_gemm(
    const __bf16* __restrict__ ctx, const __bf16* __restrict__ wot,
    const float* __restrict__ bo, float* __restrict__ out) {
  __shared__ __align__(16) __bf16 Bl[2][64][40];
  const unsigned tid = threadIdx.x;
  const unsigned wave = tid >> 5, lane = tid & 31u;
  const unsigned m0 = blockIdx.x * 128u + wave * 16u;
  const unsigned n0 = blockIdx.y * 64u;
  const unsigned nl = tid >> 2;
  const unsigned kc = (tid & 3u) * 8u;

  v8f ac[4] = {};
  cp128(&Bl[0][nl][kc], wot + (n0 + nl) * EE + kc);
  wait_cp();
  __syncthreads();

  for (unsigned step = 0; step < EE / 32u; ++step) {
    const unsigned buf = step & 1u;
    if (step + 1 < EE / 32u)
      cp128(&Bl[buf ^ 1u][nl][kc], wot + (n0 + nl) * EE + (step + 1) * 32u + kc);
    v16bf a = load_fragA(ctx, EE, m0, step * 32u);
#pragma unroll
    for (unsigned s = 0; s < 4; ++s)
      ac[s] = wmma_bf16(a, load_fragB(&Bl[buf][0][0], 40, s * 16u, 0), ac[s]);
    wait_cp();
    __syncthreads();
  }
  const unsigned hsel = lane >> 4, nlane = lane & 15u;
#pragma unroll
  for (unsigned s = 0; s < 4; ++s) {
    const unsigned col = n0 + s * 16u + nlane;
    const float b = bo[col];
#pragma unroll
    for (unsigned r = 0; r < 8; ++r)
      st_f32(out, (m0 + r + 8u * hsel) * EE + col, ac[s][r] + b);
  }
}

extern "C" void kernel_launch(void* const* d_in, const int* in_sizes, int n_in,
                              void* d_out, int out_size, void* d_ws, size_t ws_size,
                              hipStream_t stream) {
  (void)in_sizes; (void)n_in; (void)out_size; (void)ws_size;
  const float* x    = (const float*)d_in[0];
  const float* mask = (const float*)d_in[1];
  const float* Wq   = (const float*)d_in[2];
  const float* bq   = (const float*)d_in[3];
  const float* Wk   = (const float*)d_in[4];
  const float* bk   = (const float*)d_in[5];
  const float* Wv   = (const float*)d_in[6];
  const float* bv   = (const float*)d_in[7];
  const float* Wo   = (const float*)d_in[8];
  const float* bo   = (const float*)d_in[9];

  float* out   = (float*)d_out;
  float* probs = out + (size_t)RR * CC * EE;

  const size_t RCE = (size_t)RR * CC * EE;
  __bf16* xb  = (__bf16*)d_ws;        // reused as ctx after QKV
  __bf16* qb  = xb + RCE;
  __bf16* kb  = qb + RCE;
  __bf16* vb  = kb + RCE;
  __bf16* wqt = vb + RCE;
  __bf16* wkt = wqt + (size_t)EE * EE;
  __bf16* wvt = wkt + (size_t)EE * EE;
  __bf16* wot = wvt + (size_t)EE * EE;
  __bf16* ctx = xb;                   // alias: xb dead after qkv_gemm

  cvt_f32_bf16<<<4096, 256, 0, stream>>>(x, xb, (long long)RCE);
  const int wtBlocks = (EE * EE + 255) / 256;
  wt_transpose<<<wtBlocks, 256, 0, stream>>>(Wq, wqt);
  wt_transpose<<<wtBlocks, 256, 0, stream>>>(Wk, wkt);
  wt_transpose<<<wtBlocks, 256, 0, stream>>>(Wv, wvt);
  wt_transpose<<<wtBlocks, 256, 0, stream>>>(Wo, wot);

  qkv_gemm<<<dim3(RR * CC / 128, EE / 64), 256, 0, stream>>>(
      xb, wqt, wkt, wvt, bq, bk, bv, qb, kb, vb);
  attn_kernel<<<dim3(CC, HH), 256, 0, stream>>>(qb, kb, vb, mask, probs, ctx);
  out_gemm<<<dim3(RR * CC / 128, EE / 64), 256, 0, stream>>>(ctx, wot, bo, out);
}
